// LoRALinear_43250320670934
// MI455X (gfx1250) — matrix operand order repaired and checked by reading, not compiled, not run on hardware
//
#include <hip/hip_runtime.h>
#include <cstdint>
#include <cstddef>

// ---------------------------------------------------------------------------
// LoRA linear, MI455X (gfx1250): out = x @ (W + 2*(B@A))^T + bias
//   M=8192, N=4096, K=4096, rank=8.  ~275 GFLOP vs ~335 MB HBM -> compute
//   bound; bf16 WMMA 16x16x32 (8x the K/instr of the f32 16x16x4 path) with
//   f32 accumulation. f32->bf16 conversion + rank-8 LoRA weight fixup fused
//   into LDS staging. 128x128 block tile, K=32 double-buffered stages,
//   8 waves (wave32), 8 v_wmma per wave per stage.
//   Data movers: lora_B block row -> LDS via global_load_async_to_lds_b128
//   (ASYNCcnt); per-stage 8x32 strided A tile -> LDS via TDM
//   tensor_load_to_lds with an ISA 8.3/8.4 D# descriptor (TENSORcnt),
//   pipelined two stages ahead.
// ---------------------------------------------------------------------------

#define D_IN   4096
#define D_OUT  4096
#define M_TOT  8192
#define RANK   8
#define KT     32
#define NSTAGE (D_IN / KT)      // 128
#define BM     128
#define BN     128
#define XSTR   40               // 32 + 8 bf16 pad -> 80B row stride, conflict-free b128
#define LORA_SCALE 2.0f

typedef __attribute__((ext_vector_type(16))) __bf16 v16bf;
typedef __attribute__((ext_vector_type(8)))  float  v8f;
typedef __attribute__((ext_vector_type(4)))  float  f4;
typedef __attribute__((ext_vector_type(4)))  __bf16 bf4;
typedef __attribute__((ext_vector_type(4)))  int    i4v;
typedef __attribute__((ext_vector_type(8)))  int    i8v;
typedef __attribute__((ext_vector_type(4)))  unsigned int u4v;

#if defined(__has_builtin)
# if __has_builtin(__builtin_amdgcn_global_load_async_to_lds_b128) && \
     __has_builtin(__builtin_amdgcn_s_wait_asynccnt)
#  define USE_ASYNC 1
# endif
# if __has_builtin(__builtin_amdgcn_tensor_load_to_lds) && \
     __has_builtin(__builtin_amdgcn_s_wait_tensorcnt)
#  define USE_TDM 1
# endif
#endif
#ifndef USE_ASYNC
# define USE_ASYNC 0
#endif
#ifndef USE_TDM
# define USE_TDM 0
#endif

#if USE_ASYNC
// Probed prototype: (int4 as(1)* gsrc, int4 as(3)* ldst, Imm offset, Imm cpol)
typedef __attribute__((address_space(1))) i4v gi4_t;
typedef __attribute__((address_space(3))) i4v li4_t;
__device__ __forceinline__ void async_cp_b128(const void* gsrc, void* ldst) {
  __builtin_amdgcn_global_load_async_to_lds_b128((gi4_t*)gsrc, (li4_t*)ldst, 0, 0);
}
#endif

// 32-bit LDS byte offset of a __shared__ object (ptrtoint of an AS(3) pointer)
typedef __attribute__((address_space(3))) const void lvoid_c;
__device__ __forceinline__ unsigned lds_off(const void* p) {
  return (unsigned)(unsigned long long)(lvoid_c*)p;
}

__global__ __launch_bounds__(256) void lora_gemm_bf16_wmma(
    const float* __restrict__ x,     // [M_TOT, D_IN]
    const float* __restrict__ w,     // [D_OUT, D_IN]
    const float* __restrict__ la,    // [RANK, D_IN]
    const float* __restrict__ lb,    // [D_OUT, RANK]
    const float* __restrict__ bias,  // [D_OUT]
    float* __restrict__ out)         // [M_TOT, D_OUT]
{
  __shared__ __align__(16) __bf16 Xs[2][BM][XSTR];
  __shared__ __align__(16) __bf16 Ws[2][BN][XSTR];
  __shared__ __align__(16) float  As[2][RANK][KT];
  __shared__ __align__(16) float  Bs[BN][RANK];

  const int tid    = threadIdx.x;
  const int tile_m = blockIdx.y * BM;
  const int tile_n = blockIdx.x * BN;
  const int lane   = tid & 31;
  const int l16    = lane & 15;
  const int hf     = lane >> 4;     // half-wave
  const int wid    = tid >> 5;      // 0..7
  const int wm     = wid >> 2;      // 0..1 : 64 M-rows per wave
  const int wn     = wid & 3;       // 0..3 : 32 N-cols per wave

  // Per-thread (row, k) mapping for staging a 128x32 f32 chunk (4 float4 each)
  int rws[4], kqs[4];
#pragma unroll
  for (int j = 0; j < 4; ++j) {
    int f = tid + j * 256;
    rws[j] = f >> 3;
    kqs[j] = (f & 7) << 2;
  }

  auto ldXW = [&](int kt, f4* xv, f4* wv) {
#pragma unroll
    for (int j = 0; j < 4; ++j) {
      xv[j] = *(const f4*)(x + (size_t)(tile_m + rws[j]) * D_IN + kt + kqs[j]);
      wv[j] = *(const f4*)(w + (size_t)(tile_n + rws[j]) * D_IN + kt + kqs[j]);
    }
  };

  // Convert staged chunk to bf16; fold LoRA rank-8 correction into W tile.
  auto cvst = [&](int nb, const f4* xv, const f4* wv) {
#pragma unroll
    for (int j = 0; j < 4; ++j) {
      const int row = rws[j], kq = kqs[j];
      bf4 xb;
#pragma unroll
      for (int c = 0; c < 4; ++c) xb[c] = (__bf16)xv[j][c];
      *(bf4*)&Xs[nb][row][kq] = xb;

      f4 d = {0.f, 0.f, 0.f, 0.f};
#pragma unroll
      for (int r = 0; r < RANK; ++r) {
        f4 a4 = *(const f4*)&As[nb][r][kq];
        d += Bs[row][r] * a4;
      }
      f4 res = wv[j] + LORA_SCALE * d;
      bf4 wb;
#pragma unroll
      for (int c = 0; c < 4; ++c) wb[c] = (__bf16)res[c];
      *(bf4*)&Ws[nb][row][kq] = wb;
    }
  };

  // Build a 16x32 bf16 fragment (two b128 LDS reads at per-half offsets).
  auto ldfrag = [&](const __bf16* rowp, int o0, int o1) -> v16bf {
    i4v lo = *(const i4v*)(rowp + o0);
    i4v hi = *(const i4v*)(rowp + o1);
    i8v u;
    u[0] = lo[0]; u[1] = lo[1]; u[2] = lo[2]; u[3] = lo[3];
    u[4] = hi[0]; u[5] = hi[1]; u[6] = hi[2]; u[7] = hi[3];
    return __builtin_bit_cast(v16bf, u);
  };

  // ---- A-chunk staging: TDM (preferred) > async-to-LDS > plain ------------
#if USE_TDM
  // D# group1 (ISA 8.4), uniform for all stages:
  //   data_size=4B(code 2), tensor_dim0=D_IN, tensor_dim1=RANK,
  //   tile_dim0=KT, tile_dim1=RANK, tile_dim2=0, dim0_stride=D_IN.
  i8v g1;
  g1[0] = (int)(2u << 16);                                   // data_size
  g1[1] = (int)(((unsigned)D_IN & 0xFFFFu) << 16);           // tensor_dim0 lo
  g1[2] = (int)((((unsigned)D_IN >> 16) & 0xFFFFu) |
                (((unsigned)RANK & 0xFFFFu) << 16));         // dim0 hi | dim1 lo
  g1[3] = (int)(((unsigned)KT & 0xFFFFu) << 16);             // dim1 hi=0 | tile_dim0
  g1[4] = (int)((unsigned)RANK & 0xFFFFu);                   // tile_dim1 | tile_dim2=0
  g1[5] = (int)(unsigned)D_IN;                               // dim0_stride lo32
  g1[6] = 0;                                                 // stride hi | dim1_stride lo
  g1[7] = 0;
  const i4v gz = {0, 0, 0, 0};                               // groups 2/3: dims unused
  const i8v gz8 = {0, 0, 0, 0, 0, 0, 0, 0};                  // extra group (6-arg form)
  auto tdmA = [&](int stage) {     // A[0:8, stage*KT : +KT] -> As[stage&1]
    if (wid == 0) {
      const unsigned long long ga =
          (unsigned long long)(uintptr_t)(la + (size_t)stage * KT);
      u4v g0;
      g0[0] = 1u;                                            // count=1 (valid, user)
      g0[1] = lds_off(&As[stage & 1][0][0]);                 // lds_addr (bytes)
      g0[2] = (unsigned)(ga & 0xFFFFFFFFull);                // global_addr[31:0]
      g0[3] = (unsigned)((ga >> 32) & 0x01FFFFFFull) | (2u << 30); // [56:32]|type=2
      __builtin_amdgcn_tensor_load_to_lds(g0, g1, gz, gz, gz8, 0);
    }
  };
#elif USE_ASYNC
  auto asyncA = [&](int stage) {
    if (tid < 64) {
      const int r = tid >> 3, kq = (tid & 7) << 2;
      async_cp_b128(la + (size_t)r * D_IN + stage * KT + kq,
                    &As[stage & 1][r][kq]);
    }
  };
#else
  auto ldA = [&](int stage, f4& ar) {
    if (tid < 64) {
      const int r = tid >> 3, kq = (tid & 7) << 2;
      ar = *(const f4*)(la + (size_t)r * D_IN + stage * KT + kq);
    }
  };
  auto stA = [&](int stage, f4 ar) {
    if (tid < 64) {
      const int r = tid >> 3, kq = (tid & 7) << 2;
      *(f4*)&As[stage & 1][r][kq] = ar;
    }
  };
#endif

  // ---- Prologue: Bs (128x8 contiguous 4KB), A chunks 0 & 1, X/W stage 0 ----
#if USE_ASYNC
  async_cp_b128(lb + (size_t)tile_n * RANK + tid * 4,
                ((float*)&Bs[0][0]) + tid * 4);
#else
  f4 bsr = *(const f4*)(lb + (size_t)tile_n * RANK + tid * 4);
#endif
#if USE_TDM
  tdmA(0);
  tdmA(1);
#elif USE_ASYNC
  asyncA(0);
  asyncA(1);
#else
  f4 a0r, a1r;
  ldA(0, a0r);
  ldA(1, a1r);
#endif
  f4 xv[4], wv[4];
  ldXW(0, xv, wv);
#if USE_ASYNC
  __builtin_amdgcn_s_wait_asynccnt(0);
#else
  *(f4*)(((float*)&Bs[0][0]) + tid * 4) = bsr;
#endif
#if USE_TDM
  if (wid == 0) __builtin_amdgcn_s_wait_tensorcnt(0);
#elif !USE_ASYNC
  stA(0, a0r);
  stA(1, a1r);
#endif
  __syncthreads();                 // Bs, As[0], As[1] visible
  cvst(0, xv, wv);                 // stage 0 -> LDS (bf16, LoRA folded)
  __syncthreads();                 // stage 0 visible

  v8f acc[4][2];
  {
    const v8f z = {0.f, 0.f, 0.f, 0.f, 0.f, 0.f, 0.f, 0.f};
#pragma unroll
    for (int sm = 0; sm < 4; ++sm)
#pragma unroll
      for (int sn = 0; sn < 2; ++sn) acc[sm][sn] = z;
  }

  // ---- Main K loop: prefetch next stage, WMMA on current, convert, barrier --
  for (int i = 0; i < NSTAGE; ++i) {
    const int buf = i & 1;
    const bool havenext = (i + 1 < NSTAGE);
#if USE_TDM
    if (i + 2 < NSTAGE) tdmA(i + 2);
#elif USE_ASYNC
    if (i + 2 < NSTAGE) asyncA(i + 2);
#else
    f4 anr;
    if (i + 2 < NSTAGE) ldA(i + 2, anr);
#endif
    if (havenext) ldXW((i + 1) * KT, xv, wv);

    // B fragments: lanes 0-15 hold K {0-7,16-23}, lanes 16-31 hold {8-15,24-31}
    v16bf bfr[2];
#pragma unroll
    for (int sn = 0; sn < 2; ++sn)
      bfr[sn] = ldfrag(&Ws[buf][wn * 32 + sn * 16 + l16][0], hf * 8, 16 + hf * 8);

    // A fragments: lanes 0-15 hold K {0-7,16-23}, lanes 16-31 hold {8-15,24-31}
#pragma unroll
    for (int sm = 0; sm < 4; ++sm) {
      v16bf af = ldfrag(&Xs[buf][wm * 64 + sm * 16 + l16][0], hf * 8, 16 + hf * 8);
#pragma unroll
      for (int sn = 0; sn < 2; ++sn)
        acc[sm][sn] = __builtin_amdgcn_wmma_f32_16x16x32_bf16(
            false, af, false, bfr[sn], (short)0, acc[sm][sn], false, false);
    }

    if (havenext) {
#if !USE_TDM && !USE_ASYNC
      if (i + 2 < NSTAGE) stA(i + 2, anr);
#endif
      cvst((i + 1) & 1, xv, wv);
    }
#if USE_TDM
    if (wid == 0) __builtin_amdgcn_s_wait_tensorcnt(0);
#elif USE_ASYNC
    __builtin_amdgcn_s_wait_asynccnt(0);
#endif
    __syncthreads();
  }

  // ---- Epilogue: bias add + store (lane l16 -> N, VGPR r + 8*hf -> M) ------
#pragma unroll
  for (int sn = 0; sn < 2; ++sn) {
    const int n = tile_n + wn * 32 + sn * 16 + l16;
    const float bv = bias[n];
#pragma unroll
    for (int sm = 0; sm < 4; ++sm) {
      const int mb = tile_m + wm * 64 + sm * 16 + hf * 8;
#pragma unroll
      for (int r = 0; r < 8; ++r)
        out[(size_t)(mb + r) * D_OUT + n] = acc[sm][sn][r] + bv;
    }
  }
}

extern "C" void kernel_launch(void* const* d_in, const int* in_sizes, int n_in,
                              void* d_out, int out_size, void* d_ws, size_t ws_size,
                              hipStream_t stream) {
  const float* x  = (const float*)d_in[0];   // [4,2048,4096]
  const float* w  = (const float*)d_in[1];   // [4096,4096]
  const float* la = (const float*)d_in[2];   // [8,4096]
  const float* lb = (const float*)d_in[3];   // [4096,8]
  const float* bs = (const float*)d_in[4];   // [4096]
  float* out = (float*)d_out;                // [4,2048,4096]

  dim3 grid(D_OUT / BN, M_TOT / BM);         // 32 x 64
  lora_gemm_bf16_wmma<<<grid, 256, 0, stream>>>(x, w, la, lb, bs, out);

  (void)in_sizes; (void)n_in; (void)out_size; (void)d_ws; (void)ws_size;
}
